// DynKQAE2_33389075759177
// MI455X (gfx1250) — matrix operand order repaired and checked by reading, not compiled, not run on hardware
//
#include <hip/hip_runtime.h>
#include <type_traits>

// ---------------------------------------------------------------------------
// DynKQAE2 forward for gfx1250 (MI455X): f16 WMMA GEMM chain + wave32 top-k.
// ---------------------------------------------------------------------------

typedef __attribute__((ext_vector_type(16))) _Float16 v16h;
typedef __attribute__((ext_vector_type(8)))  float    v8f;
typedef __attribute__((ext_vector_type(4)))  float    f4;
typedef __attribute__((ext_vector_type(4)))  _Float16 h4;

union Frag16 { unsigned u[8]; v16h h; };

__device__ __forceinline__ int wave_sum_i32(int v) {
  #pragma unroll
  for (int o = 16; o > 0; o >>= 1) v += __shfl_xor(v, o, 32);
  return v;
}
__device__ __forceinline__ float wave_sum_f32(float v) {
  #pragma unroll
  for (int o = 16; o > 0; o >>= 1) v += __shfl_xor(v, o, 32);
  return v;
}

// ---------------------------------------------------------------------------
// Software-pipelined, LDS double-buffered GEMM:
//   C[M,N] = act(A[M,K] * W[N,K]^T + bias [+ Cin]), optional per-row A scale.
// REQUIRES: M % 128 == 0, N % 128 == 0, K % 32 == 0 (true for all launches).
// Block tile 128(M) x 128(N), K-step 32, 256 threads = 8 waves (wave32).
// Each wave computes a 32x64 sub-tile = 2x4 v_wmma_f32_16x16x32_f16.
// Pipeline: global loads for tile t+1 are issued BEFORE the WMMA block of
// tile t; the loadcnt wait lands in the convert/LDS-store phase afterwards.
// ---------------------------------------------------------------------------
template <typename TA, typename TC, bool HAS_SCALE, bool HAS_CIN>
__global__ __launch_bounds__(256) void gemm_wmma_f16(
    const TA* __restrict__ A, int lda,
    const float* __restrict__ W, int ldw,
    const float* __restrict__ bias,       // may be nullptr
    const float* __restrict__ rowscale,   // used iff HAS_SCALE
    const TC* __restrict__ Cin,           // used iff HAS_CIN
    TC* __restrict__ C, int ldc,
    int M, int N, int K, int relu)
{
  // 32 K-cols padded to 40 halves; two buffers for the pipeline
  __shared__ __align__(16) _Float16 sA[2][128 * 40];
  __shared__ __align__(16) _Float16 sB[2][128 * 40];

  const int tid    = threadIdx.x;
  const int lane   = tid & 31;
  const int wave   = tid >> 5;
  const int wm     = wave & 3;        // M quadrant (4 x 32 rows)
  const int wn     = wave >> 2;       // N half     (2 x 64 cols)
  const int mBlk   = blockIdx.y * 128;
  const int nBlk   = blockIdx.x * 128;
  const int halfId = lane >> 4;       // 0: lanes 0-15, 1: lanes 16-31
  const int l16    = lane & 15;

  // staging geometry: 8 threads per row, 4 consecutive elements per thread,
  // each thread covers rows (arow + 32*i), i = 0..3, for both A and W tiles.
  const int arow = tid >> 3;          // 0..31
  const int acol = (tid & 7) * 4;     // 0..28

  using AV = std::conditional_t<sizeof(TA) == 4, f4, h4>;

  const TA*    Abase = A + (long)(mBlk + arow) * lda + acol;
  const float* Wbase = W + (long)(nBlk + arow) * ldw + acol;

  float rs[4];
  #pragma unroll
  for (int i = 0; i < 4; ++i)
    rs[i] = HAS_SCALE ? rowscale[mBlk + arow + 32 * i] : 1.f;

  AV ra[4];
  f4 rw[4];

  auto issue_loads = [&](int k0) {
    #pragma unroll
    for (int i = 0; i < 4; ++i)
      ra[i] = *(const AV*)(Abase + (long)(32 * i) * lda + k0);
    #pragma unroll
    for (int i = 0; i < 4; ++i)
      rw[i] = *(const f4*)(Wbase + (long)(32 * i) * ldw + k0);
  };
  auto store_tiles = [&](int buf) {
    #pragma unroll
    for (int i = 0; i < 4; ++i) {
      int r = arow + 32 * i;
      h4 hv;
      if constexpr (sizeof(TA) == 4) {
        f4 v = ra[i];
        if (HAS_SCALE) v *= rs[i];
        hv[0] = (_Float16)v[0]; hv[1] = (_Float16)v[1];
        hv[2] = (_Float16)v[2]; hv[3] = (_Float16)v[3];
      } else {
        hv = ra[i];
        if (HAS_SCALE) {
          #pragma unroll
          for (int j = 0; j < 4; ++j) hv[j] = (_Float16)((float)hv[j] * rs[i]);
        }
      }
      *(h4*)&sA[buf][r * 40 + acol] = hv;
      f4 w = rw[i];
      h4 hw;
      hw[0] = (_Float16)w[0]; hw[1] = (_Float16)w[1];
      hw[2] = (_Float16)w[2]; hw[3] = (_Float16)w[3];
      *(h4*)&sB[buf][r * 40 + acol] = hw;
    }
  };

  v8f acc[2][4] = {};

  const int nk = K >> 5;              // K / 32 tiles
  issue_loads(0);
  store_tiles(0);

  for (int kt = 0; kt < nk; ++kt) {
    const int buf = kt & 1;
    const bool more = (kt + 1) < nk;
    if (more) issue_loads((kt + 1) << 5);            // overlap with WMMAs
    if ((kt + 2) < nk)                               // guarded far prefetch
      __builtin_prefetch(Abase + ((long)(kt + 2) << 5), 0, 3);
    __syncthreads();                                 // sA/sB[buf] visible

    // ---- gather WMMA fragments from LDS per ISA 7.12.2 wave32 layouts ----
    Frag16 a[2], b[4];
    const unsigned* pA = (const unsigned*)sA[buf];   // u32 row stride = 20
    const unsigned* pB = (const unsigned*)sB[buf];
    #pragma unroll
    for (int s = 0; s < 2; ++s) {
      // A 16x32 f16: lane half 0 -> K{0-7,16-23}, half 1 -> K{8-15,24-31}
      int baseA = (wm * 32 + s * 16 + l16) * 20 + halfId * 4;
      #pragma unroll
      for (int v = 0; v < 4; ++v) a[s].u[v]     = pA[baseA + v];
      #pragma unroll
      for (int v = 0; v < 4; ++v) a[s].u[4 + v] = pA[baseA + 8 + v];
    }
    #pragma unroll
    for (int j = 0; j < 4; ++j) {
      // B 32x16 f16: lane half 0 -> K{0-7,16-23}, half 1 -> K{8-15,24-31}, col = l16
      int baseB = (wn * 64 + j * 16 + l16) * 20 + halfId * 4;
      #pragma unroll
      for (int v = 0; v < 4; ++v) b[j].u[v]     = pB[baseB + v];
      #pragma unroll
      for (int v = 0; v < 4; ++v) b[j].u[4 + v] = pB[baseB + 8 + v];
    }

    #pragma unroll
    for (int i = 0; i < 2; ++i)
      #pragma unroll
      for (int j = 0; j < 4; ++j)
        acc[i][j] = __builtin_amdgcn_wmma_f32_16x16x32_f16(
            false, a[i].h, false, b[j].h, (short)0, acc[i][j], false, false);

    if (more) store_tiles(buf ^ 1);   // waits for the early loads here
  }

  // ---- epilogue: C/D layout -> m = vgpr + 8*halfId, n = l16 ----
  #pragma unroll
  for (int i = 0; i < 2; ++i) {
    #pragma unroll
    for (int j = 0; j < 4; ++j) {
      int n = nBlk + wn * 64 + j * 16 + l16;
      float bv = bias ? bias[n] : 0.f;
      #pragma unroll
      for (int v = 0; v < 8; ++v) {
        int m = mBlk + wm * 32 + i * 16 + halfId * 8 + v;
        long idx = (long)m * ldc + n;
        float val = acc[i][j][v] + bv;
        if (HAS_CIN) val += (float)Cin[idx];
        if (relu) val = fmaxf(val, 0.f);
        C[idx] = (TC)val;
      }
    }
  }
}

// ---------------------------------------------------------------------------
// k-predictor tail: k = clip(sigmoid(h2.Wk3 + b)*1024 * sigmoid(ks)*2, 1,1024)
// One wave per row. Writes k (output) and 1/k (scratch, GEMM row-scale).
// ---------------------------------------------------------------------------
__global__ __launch_bounds__(256) void kpred_kernel(
    const _Float16* __restrict__ h2, const float* __restrict__ Wk3,
    const float* __restrict__ bk3, const float* __restrict__ kscale,
    float* __restrict__ invk, float* __restrict__ kout,
    float* __restrict__ zero_slot, int B)
{
  int lane = threadIdx.x & 31;
  int row  = blockIdx.x * 8 + (threadIdx.x >> 5);
  if (blockIdx.x == 0 && threadIdx.x == 0) zero_slot[0] = 0.f;
  if (row >= B) return;
  float s = 0.f;
  #pragma unroll
  for (int i = 0; i < 16; ++i) {
    int c = i * 32 + lane;
    s += (float)h2[(long)row * 512 + c] * Wk3[c];
  }
  s = wave_sum_f32(s);
  if (lane == 0) {
    float kraw = 1024.f / (1.f + __expf(-(s + bk3[0])));
    float ks   = 1.f / (1.f + __expf(-kscale[0]));
    float kc   = kraw * ks * 2.f;
    kc = kc < 1.f ? 1.f : (kc > 1024.f ? 1024.f : kc);
    kout[row] = kc;
    invk[row] = 1.f / kc;
  }
}

// ---------------------------------------------------------------------------
// Dynamic top-k mask: khot[i,j] = (rank of logits[i,j] desc, ties by index) < k.
// One wave per row; lane owns 32 contiguous keys. 32-step bitwise binary
// search for the ki-th largest order-preserving uint key, exact tie handling
// via wave prefix scan.
// ---------------------------------------------------------------------------
__global__ __launch_bounds__(256) void topk_kernel(
    const _Float16* __restrict__ logits, const float* __restrict__ kvals,
    float* __restrict__ khot, int B)
{
  int lane = threadIdx.x & 31;
  int row  = blockIdx.x * 8 + (threadIdx.x >> 5);
  if (row >= B) return;

  const _Float16* lp = logits + (long)row * 1024 + lane * 32;
  unsigned key[32];
  #pragma unroll
  for (int i = 0; i < 32; ++i) {
    unsigned u = __float_as_uint((float)lp[i]);
    key[i] = (u & 0x80000000u) ? ~u : (u | 0x80000000u);  // order-preserving
  }

  float kc = kvals[row];
  int ki = (int)ceilf(kc);
  ki = ki < 1 ? 1 : (ki > 1024 ? 1024 : ki);

  // T = ki-th largest key: max T with count(key >= T) >= ki
  unsigned T = 0;
  for (int bit = 31; bit >= 0; --bit) {
    unsigned trial = T | (1u << bit);
    int c = 0;
    #pragma unroll
    for (int i = 0; i < 32; ++i) c += (key[i] >= trial) ? 1 : 0;
    c = wave_sum_i32(c);
    if (c >= ki) T = trial;
  }

  int g = 0, e = 0;
  #pragma unroll
  for (int i = 0; i < 32; ++i) { g += key[i] > T; e += key[i] == T; }
  int gtot = wave_sum_i32(g);
  int Eneed = ki - gtot;            // how many ties (lowest index) to accept

  // exclusive prefix of tie counts across lanes (index order = lane order)
  int inc = e;
  #pragma unroll
  for (int o = 1; o < 32; o <<= 1) {
    int y = __shfl_up(inc, o, 32);
    if (lane >= o) inc += y;
  }
  int run = inc - e;

  float4* out4 = (float4*)(khot + (long)row * 1024 + lane * 32);
  #pragma unroll
  for (int bq = 0; bq < 8; ++bq) {
    float4 v;
    #pragma unroll
    for (int t = 0; t < 4; ++t) {
      int i = bq * 4 + t;
      bool eq  = (key[i] == T);
      bool sel = (key[i] > T) || (eq && run < Eneed);
      if (eq) run++;
      ((float*)&v)[t] = sel ? 1.f : 0.f;
    }
    out4[bq] = v;
  }
}

// ---------------------------------------------------------------------------
extern "C" void kernel_launch(void* const* d_in, const int* in_sizes, int n_in,
                              void* d_out, int out_size, void* d_ws, size_t ws_size,
                              hipStream_t stream) {
  const float* x     = (const float*)d_in[0];
  const float* W_e1  = (const float*)d_in[1];
  const float* b_e1  = (const float*)d_in[2];
  const float* W_e2  = (const float*)d_in[3];
  const float* b_e2  = (const float*)d_in[4];
  const float* W_cb  = (const float*)d_in[5];
  const float* W_d1  = (const float*)d_in[6];
  const float* b_d1  = (const float*)d_in[7];
  const float* W_d2  = (const float*)d_in[8];
  const float* b_d2  = (const float*)d_in[9];
  const float* W_k1  = (const float*)d_in[10];
  const float* b_k1  = (const float*)d_in[11];
  const float* W_k2  = (const float*)d_in[12];
  const float* b_k2  = (const float*)d_in[13];
  const float* W_k3  = (const float*)d_in[14];
  const float* b_k3  = (const float*)d_in[15];
  const float* kscal = (const float*)d_in[16];

  const int B = in_sizes[0] / 256;   // 32768

  // workspace layout (f16 activations): logits | bufA | bufB | invk
  char* ws = (char*)d_ws;
  _Float16* logits = (_Float16*)ws;                                   // B*1024
  _Float16* bufA   = (_Float16*)(ws + (size_t)B * 1024 * 2);          // B*512
  _Float16* bufB   = (_Float16*)(ws + (size_t)B * 1024 * 2 + (size_t)B * 512 * 2);
  float*    invk   = (float*)   (ws + (size_t)B * 1024 * 2 + (size_t)B * 512 * 4);

  // output layout: recon [B,256] | khot [B,1024] | 0.0 | k [B]
  float* recon = (float*)d_out;
  float* khot  = recon + (size_t)B * 256;
  float* zero  = khot + (size_t)B * 1024;
  float* kout  = zero + 1;

  dim3 blk(256);
  int gy = B / 128;

  // 1) h1 = relu(x @ W_e1^T + b_e1)                [B,512]
  gemm_wmma_f16<float, _Float16, false, false><<<dim3(4, gy), blk, 0, stream>>>(
      x, 256, W_e1, 256, b_e1, nullptr, nullptr, bufA, 512, B, 512, 256, 1);
  // 2) logits = h1 @ W_e2^T + b_e2                 [B,1024]
  gemm_wmma_f16<_Float16, _Float16, false, false><<<dim3(8, gy), blk, 0, stream>>>(
      bufA, 512, W_e2, 512, b_e2, nullptr, nullptr, logits, 1024, B, 1024, 512, 0);
  // 3a) partial = x @ W_k1[:, :256]^T              [B,512]
  gemm_wmma_f16<float, _Float16, false, false><<<dim3(4, gy), blk, 0, stream>>>(
      x, 256, W_k1, 1280, nullptr, nullptr, nullptr, bufB, 512, B, 512, 256, 0);
  // 3b) hk = relu(partial + logits @ W_k1[:,256:]^T + b_k1)
  gemm_wmma_f16<_Float16, _Float16, false, true><<<dim3(4, gy), blk, 0, stream>>>(
      logits, 1024, W_k1 + 256, 1280, b_k1, nullptr, bufB, bufA, 512, B, 512, 1024, 1);
  // 4) h2 = relu(hk @ W_k2^T + b_k2)               [B,512]
  gemm_wmma_f16<_Float16, _Float16, false, false><<<dim3(4, gy), blk, 0, stream>>>(
      bufA, 512, W_k2, 512, b_k2, nullptr, nullptr, bufB, 512, B, 512, 512, 1);
  // 5) k = clip(sigmoid(h2.Wk3+b)*1024 * sigmoid(ks)*2, 1, 1024); also 1/k, 0.0
  kpred_kernel<<<dim3(B / 8), blk, 0, stream>>>(bufB, W_k3, b_k3, kscal,
                                                invk, kout, zero, B);
  // 6) khot = (rank(logits) < k)                   [B,1024]
  topk_kernel<<<dim3(B / 8), blk, 0, stream>>>(logits, kout, khot, B);
  // 7) q = (khot * 1/k) @ W_cb^T                   [B,256]
  gemm_wmma_f16<float, _Float16, true, false><<<dim3(2, gy), blk, 0, stream>>>(
      khot, 1024, W_cb, 1024, nullptr, invk, nullptr, bufB, 256, B, 256, 1024, 0);
  // 8) hd = relu(q @ W_d1^T + b_d1)                [B,512]
  gemm_wmma_f16<_Float16, _Float16, false, false><<<dim3(4, gy), blk, 0, stream>>>(
      bufB, 256, W_d1, 256, b_d1, nullptr, nullptr, bufA, 512, B, 512, 256, 1);
  // 9) recon = hd @ W_d2^T + b_d2                  [B,256] (f32 out)
  gemm_wmma_f16<_Float16, float, false, false><<<dim3(2, gy), blk, 0, stream>>>(
      bufA, 512, W_d2, 512, b_d2, nullptr, nullptr, recon, 256, B, 256, 512, 0);
}
